// MoEConformerEncoder_33990371181314
// MI455X (gfx1250) — matrix operand order repaired and checked by reading, not compiled, not run on hardware
//
#include <hip/hip_runtime.h>
#include <stdint.h>

// ---------------------------------------------------------------------------
// MoE Conformer encoder for MI455X (gfx1250, wave32, WMMA).
// All GEMMs (conv-as-GEMM, QKV, scores, PV, out-proj, FF, experts) run on
// v_wmma_f32_16x16x32_f16. A-side activations are kept in f16 end-to-end,
// weights pre-packed to f16 K-panels, f32 accumulation. Wave tile 32x64
// (2 A-frag x 4 B-frag -> 8 WMMAs / 32-deep k-step). The 4 waves of a block
// stack along M and share B: B K-panels are staged into LDS (async-to-LDS on
// toolchains that expose it, sync fallback otherwise) and read via ds loads.
// ---------------------------------------------------------------------------

typedef __attribute__((ext_vector_type(16))) _Float16 v16h;
typedef __attribute__((ext_vector_type(8)))  _Float16 v8h;
typedef __attribute__((ext_vector_type(4)))  _Float16 v4h;
typedef __attribute__((ext_vector_type(8)))  float    v8f;
typedef __attribute__((ext_vector_type(4)))  int      v4i;

#define DEV __device__ __forceinline__

// Problem constants (fixed by the reference)
#define BB   4
#define SS   1024
#define DD   512
#define HH   8
#define HDIM 64
#define FFD  2048
#define GG   4
#define EE   2
#define HIDD 1024
#define MR   (BB*SS)          // 4096 token rows
#define TAPS 31
#define PAD  15
#define KCH  4                // B K-panels staged into LDS per barrier (16 KB)

#if defined(__has_builtin)
#if __has_builtin(__builtin_amdgcn_global_load_async_to_lds_b128)
#define ASYNC_LDS 1
#endif
#endif

#define AS1 __attribute__((address_space(1)))
#define AS3 __attribute__((address_space(3)))

// Stage 16B global -> LDS. Async path uses the CDNA5 TA async-to-LDS op
// (tracked by ASYNCcnt); fallback is a synchronous copy through VGPRs.
// Builtin signature (from probe): (AS1 v4i*, AS3 v4i*, imm offset, imm cpol).
DEV void stage_b128(const _Float16* gp, _Float16* lp) {
#ifdef ASYNC_LDS
  __builtin_amdgcn_global_load_async_to_lds_b128(
      (AS1 v4i*)gp, (AS3 v4i*)lp, 0, 0);
#else
  *(v8h*)lp = *(const v8h*)gp;
#endif
}
DEV void stage_wait() {
#ifdef ASYNC_LDS
  asm volatile("s_wait_asynccnt 0" ::: "memory");
#endif
}

DEV v8f wmma_16x16x32(v16h a, v16h b, v8f c) {
  // 8 args: (neg_a, A, neg_b, B, c_mod, C, reuse_a, reuse_b)
  return __builtin_amdgcn_wmma_f32_16x16x32_f16(false, a, false, b, (short)0, c,
                                                false, false);
}

// A fragment (16x32 f16) from f16 row-major storage; p points at row + k0.
// lane<16 holds K{0..15}, lane>=16 holds K{16..31} (matches B fragment order).
DEV v16h load_ah(const _Float16* p, int half16) {
  v8h x0 = *(const v8h*)(p + half16 * 16);
  v8h x1 = *(const v8h*)(p + half16 * 16 + 8);
  v16h a;
#pragma unroll
  for (int i = 0; i < 8; ++i) { a[i] = x0[i]; a[8 + i] = x1[i]; }
  return a;
}

// B fragment (32x16 f16): lane n reads 16 contiguous halfs; lanes 0..15 hold
// K=0..15 of column n, lanes 16..31 hold K=16..31. Works on global or LDS ptr.
DEV v16h load_b_frag(const _Float16* p) {
  v8h b0 = *(const v8h*)(p);
  v8h b1 = *(const v8h*)(p + 8);
  v16h b;
#pragma unroll
  for (int i = 0; i < 8; ++i) { b[i] = b0[i]; b[8 + i] = b1[i]; }
  return b;
}

DEV float gelu_f(float x) {   // tanh approximation (jax.nn.gelu default)
  float x3 = x * x * x;
  return 0.5f * x * (1.f + tanhf(0.7978845608028654f * (x + 0.044715f * x3)));
}

// ---------------------------------------------------------------------------
// Generic batched WMMA GEMM:  C = beta*C + res + alpha*act(A*B + bias)
// A: f16 [M,K] (lda).  Bp: f16 panels Bp[(kb*N+n)*32+kk]=B[kb*32+kk][n].
// Block = 4 waves stacked along M (128x64 tile); B staged via LDS.
// Grids are exactly divisible (no partial tiles). outHalf: store f16.
// ---------------------------------------------------------------------------
__global__ void gemm_wmma_kernel(const _Float16* __restrict__ Abase,
                                 const _Float16* __restrict__ Bbase,
                                 const float* __restrict__ bias,
                                 const float* __restrict__ res,
                                 void* __restrict__ Cbase,
                                 int M, int N, int Kdim, int lda, int ldc,
                                 int64_t a_sb, int64_t a_sh, int64_t b_sz,
                                 int64_t c_sb, int64_t c_sh, int Hdim,
                                 float alpha, int act, int accum, int outHalf) {
  __shared__ _Float16 Bsh[KCH * 64 * 32];          // 16 KB
  int z  = blockIdx.z;
  int zb = z / Hdim, zh = z - zb * Hdim;
  const _Float16* A  = Abase + zb * a_sb + zh * a_sh;
  const _Float16* Bp = Bbase + (int64_t)z * b_sz;
  int64_t coff = zb * c_sb + zh * c_sh;

  int lane = threadIdx.x, wave = threadIdx.y;
  int tid = wave * 32 + lane;
  int tileN = blockIdx.x * 64;
  int tileM = (blockIdx.y * 4 + wave) * 32;
  int half16 = lane >> 4, l15 = lane & 15;

  const _Float16* Ar0 = A + (size_t)(tileM + l15) * lda;
  const _Float16* Ar1 = Ar0 + (size_t)16 * lda;

  v8f zero8 = {0.f,0.f,0.f,0.f,0.f,0.f,0.f,0.f};
  v8f acc[8];
#pragma unroll
  for (int i = 0; i < 8; ++i) acc[i] = zero8;

  int KB = Kdim >> 5;
  for (int kb0 = 0; kb0 < KB; kb0 += KCH) {
    int kcur = KB - kb0; if (kcur > KCH) kcur = KCH;
    int nseg = kcur << 8;                          // 16B segments in chunk
#pragma unroll
    for (int i = 0; i < 8; ++i) {
      int seg = tid + (i << 7);
      if (seg < nseg) {
        int kb_l = seg >> 8, within = seg & 255;
        const _Float16* gp =
            Bp + ((size_t)(kb0 + kb_l) * N + tileN) * 32 + within * 8;
        stage_b128(gp, &Bsh[(kb_l << 11) + within * 8]);
      }
    }
    stage_wait();
    __syncthreads();
    for (int kb_l = 0; kb_l < kcur; ++kb_l) {
      int kb = kb0 + kb_l;
      if (kb + 1 < KB) {
        __builtin_prefetch(Ar0 + (kb + 1) * 32, 0, 1);   // global_prefetch_b8
        __builtin_prefetch(Ar1 + (kb + 1) * 32, 0, 1);
      }
      v16h a0 = load_ah(Ar0 + kb * 32, half16);
      v16h a1 = load_ah(Ar1 + kb * 32, half16);
      const _Float16* bs = &Bsh[(kb_l << 11)] + (size_t)l15 * 32 + half16 * 16;
      v16h b0 = load_b_frag(bs);
      v16h b1 = load_b_frag(bs + 16 * 32);
      v16h b2 = load_b_frag(bs + 32 * 32);
      v16h b3 = load_b_frag(bs + 48 * 32);
      acc[0] = wmma_16x16x32(a0, b0, acc[0]);
      acc[1] = wmma_16x16x32(a0, b1, acc[1]);
      acc[2] = wmma_16x16x32(a0, b2, acc[2]);
      acc[3] = wmma_16x16x32(a0, b3, acc[3]);
      acc[4] = wmma_16x16x32(a1, b0, acc[4]);
      acc[5] = wmma_16x16x32(a1, b1, acc[5]);
      acc[6] = wmma_16x16x32(a1, b2, acc[6]);
      acc[7] = wmma_16x16x32(a1, b3, acc[7]);
    }
    __syncthreads();
  }

  float*    Cf = (float*)Cbase + coff;
  _Float16* Ch = (_Float16*)Cbase + coff;
  const float* R = res ? res + coff : nullptr;
#pragma unroll
  for (int mt = 0; mt < 2; ++mt) {
#pragma unroll
    for (int j = 0; j < 4; ++j) {
      v8f av = acc[mt * 4 + j];
      int col = tileN + j * 16 + l15;
      float bv = bias ? bias[col] : 0.f;
#pragma unroll
      for (int r = 0; r < 8; ++r) {
        int row = tileM + mt * 16 + half16 * 8 + r;
        float t = av[r] + bv;
        if (act == 1) t = gelu_f(t);
        t *= alpha;
        size_t idx = (size_t)row * ldc + col;
        if (outHalf) {
          Ch[idx] = (_Float16)t;
        } else {
          float o = t;
          if (accum) o += Cf[idx];
          if (R)     o += R[idx];
          Cf[idx] = o;
        }
      }
    }
  }
}

// ---------------------------------------------------------------------------
// Conv1d (K=31, SAME) as shifted WMMA GEMM over [MR, DD]; A in f16 (LN1 out),
// fused bias + GELU + f32 residual. Flattened 496-panel loop with the same
// LDS B staging; per-row zero padding at sequence boundaries.
// ---------------------------------------------------------------------------
__global__ void conv_gemm_wmma_kernel(const _Float16* __restrict__ X,
                                      const _Float16* __restrict__ Wp,
                                      const float* __restrict__ bias,
                                      const float* __restrict__ res,
                                      float* __restrict__ C) {
  __shared__ _Float16 Bsh[KCH * 64 * 32];          // 16 KB
  int lane = threadIdx.x, wave = threadIdx.y;
  int tid = wave * 32 + lane;
  int tileN = blockIdx.x * 64;
  int tileM = (blockIdx.y * 4 + wave) * 32;
  int half16 = lane >> 4, l15 = lane & 15;
  int m0 = tileM + l15;
  int bb = tileM >> 10;                 // batch (32-row tile never crosses)
  int s0 = m0 & (SS - 1);
  int s1 = s0 + 16;

  v8f zero8 = {0.f,0.f,0.f,0.f,0.f,0.f,0.f,0.f};
  v8f acc[8];
#pragma unroll
  for (int i = 0; i < 8; ++i) acc[i] = zero8;
  v16h az;
#pragma unroll
  for (int i = 0; i < 16; ++i) az[i] = (_Float16)0.f;

  const int KBTOT = TAPS * (DD / 32);   // 496, multiple of KCH
  for (int kk0 = 0; kk0 < KBTOT; kk0 += KCH) {
#pragma unroll
    for (int i = 0; i < 8; ++i) {
      int seg = tid + (i << 7);                    // 0..1023
      int kb_l = seg >> 8, within = seg & 255;
      const _Float16* gp =
          Wp + ((size_t)(kk0 + kb_l) * DD + tileN) * 32 + within * 8;
      stage_b128(gp, &Bsh[(kb_l << 11) + within * 8]);
    }
    stage_wait();
    __syncthreads();
    for (int l = 0; l < KCH; ++l) {
      int kk = kk0 + l;
      int tap = kk >> 4, kb = kk & 15;
      int sp0 = s0 + tap - PAD, sp1 = s1 + tap - PAD;
      bool va0 = (unsigned)sp0 < (unsigned)SS;
      bool va1 = (unsigned)sp1 < (unsigned)SS;
      const _Float16* Ar0 =
          X + ((size_t)bb * SS + (va0 ? sp0 : 0)) * DD + kb * 32;
      const _Float16* Ar1 =
          X + ((size_t)bb * SS + (va1 ? sp1 : 0)) * DD + kb * 32;
      v16h a0 = va0 ? load_ah(Ar0, half16) : az;
      v16h a1 = va1 ? load_ah(Ar1, half16) : az;
      const _Float16* bs = &Bsh[(l << 11)] + (size_t)l15 * 32 + half16 * 16;
      v16h b0 = load_b_frag(bs);
      v16h b1 = load_b_frag(bs + 16 * 32);
      v16h b2 = load_b_frag(bs + 32 * 32);
      v16h b3 = load_b_frag(bs + 48 * 32);
      acc[0] = wmma_16x16x32(a0, b0, acc[0]);
      acc[1] = wmma_16x16x32(a0, b1, acc[1]);
      acc[2] = wmma_16x16x32(a0, b2, acc[2]);
      acc[3] = wmma_16x16x32(a0, b3, acc[3]);
      acc[4] = wmma_16x16x32(a1, b0, acc[4]);
      acc[5] = wmma_16x16x32(a1, b1, acc[5]);
      acc[6] = wmma_16x16x32(a1, b2, acc[6]);
      acc[7] = wmma_16x16x32(a1, b3, acc[7]);
    }
    __syncthreads();
  }

#pragma unroll
  for (int mt = 0; mt < 2; ++mt) {
#pragma unroll
    for (int j = 0; j < 4; ++j) {
      v8f av = acc[mt * 4 + j];
      int col = tileN + j * 16 + l15;
      float bv = bias[col];
#pragma unroll
      for (int r = 0; r < 8; ++r) {
        int row = tileM + mt * 16 + half16 * 8 + r;
        float t = gelu_f(av[r] + bv);
        size_t idx = (size_t)row * DD + col;
        C[idx] = t + res[idx];
      }
    }
  }
}

// ---------------------------------------------------------------------------
// Weight pack: W f32 [Kdim,N] row-major -> f16 panels (one thread per (kb,n)).
// ---------------------------------------------------------------------------
__global__ void pack_b_kernel(const float* __restrict__ W,
                              _Float16* __restrict__ Bp, int Kdim, int N) {
  int64_t t = (int64_t)blockIdx.x * blockDim.x + threadIdx.x;
  int64_t total = (int64_t)(Kdim >> 5) * N;
  if (t >= total) return;
  int n  = (int)(t % N);
  int kb = (int)(t / N);
  const float* src = W + (size_t)kb * 32 * N + n;
  _Float16* dst = Bp + t * 32;
#pragma unroll 4
  for (int kk = 0; kk < 32; ++kk) dst[kk] = (_Float16)src[(size_t)kk * N];
}

// K^T pack for scores (f16 in): Bp[z][(kb*SS+s)*32+kk] = K[b][s][h*64+kb*32+kk]
__global__ void pack_kT_kernel(const _Float16* __restrict__ Ksrc,
                               _Float16* __restrict__ Bp) {
  int z = blockIdx.y; int b = z / HH, h = z - b * HH;
  int t = blockIdx.x * blockDim.x + threadIdx.x;     // (HDIM/32)*SS = 2048
  if (t >= (HDIM / 32) * SS) return;
  int s  = t & (SS - 1);
  int kb = t >> 10;
  const _Float16* src = Ksrc + ((size_t)b * SS + s) * DD + h * HDIM + kb * 32;
  _Float16* dst = Bp + ((size_t)z * (HDIM / 32) * SS + t) * 32;
#pragma unroll
  for (int u = 0; u < 4; ++u)
    *(v8h*)(dst + u * 8) = *(const v8h*)(src + u * 8);
}

// V pack for PV (f16 in): Bp[z][(kb*HDIM+n)*32+kk] = V[b][kb*32+kk][h*64+n]
__global__ void pack_v_kernel(const _Float16* __restrict__ Vsrc,
                              _Float16* __restrict__ Bp) {
  int z = blockIdx.y; int b = z / HH, h = z - b * HH;
  int t = blockIdx.x * blockDim.x + threadIdx.x;     // (SS/32)*HDIM = 2048
  if (t >= (SS / 32) * HDIM) return;
  int n  = t & (HDIM - 1);
  int kb = t >> 6;
  const _Float16* src = Vsrc + ((size_t)b * SS + (size_t)kb * 32) * DD + h * HDIM + n;
  _Float16* dst = Bp + ((size_t)z * (SS / 32) * HDIM + t) * 32;
#pragma unroll 4
  for (int kk = 0; kk < 32; ++kk) dst[kk] = src[(size_t)kk * DD];
}

// f32 -> f16 convert (rows of 4)
__global__ void cvt_f32_f16_kernel(const float* __restrict__ x,
                                   _Float16* __restrict__ y) {
  int64_t i = (int64_t)blockIdx.x * blockDim.x + threadIdx.x;
  float4 v = *(const float4*)(x + i * 4);
  v4h o;
  o[0] = (_Float16)v.x; o[1] = (_Float16)v.y;
  o[2] = (_Float16)v.z; o[3] = (_Float16)v.w;
  *(v4h*)(y + i * 4) = o;
}

// ---------------------------------------------------------------------------
// LayerNorm over D=512 -> f16 output (one 256-thread block per row).
// ---------------------------------------------------------------------------
__global__ void layernorm_kernel(const float* __restrict__ x,
                                 const float* __restrict__ sc,
                                 const float* __restrict__ bi,
                                 _Float16* __restrict__ y) {
  __shared__ float red[8];
  int row = blockIdx.x, tid = threadIdx.x;
  const float* xr = x + (size_t)row * DD;
  float2 v = *(const float2*)(xr + tid * 2);
  float s = v.x + v.y;
#pragma unroll
  for (int o = 16; o; o >>= 1) s += __shfl_xor(s, o, 32);
  if ((tid & 31) == 0) red[tid >> 5] = s;
  __syncthreads();
  float mean = 0.f;
#pragma unroll
  for (int i = 0; i < 8; ++i) mean += red[i];
  mean *= (1.f / DD);
  __syncthreads();
  float d0 = v.x - mean, d1 = v.y - mean;
  float vs = d0 * d0 + d1 * d1;
#pragma unroll
  for (int o = 16; o; o >>= 1) vs += __shfl_xor(vs, o, 32);
  if ((tid & 31) == 0) red[tid >> 5] = vs;
  __syncthreads();
  float var = 0.f;
#pragma unroll
  for (int i = 0; i < 8; ++i) var += red[i];
  var *= (1.f / DD);
  float inv = rsqrtf(var + 1e-6f);
  _Float16* yr = y + (size_t)row * DD;
  yr[tid * 2]     = (_Float16)(d0 * inv * sc[tid * 2]     + bi[tid * 2]);
  yr[tid * 2 + 1] = (_Float16)(d1 * inv * sc[tid * 2 + 1] + bi[tid * 2 + 1]);
}

// Row softmax over S=1024: f32 scores in, f16 probabilities out.
__global__ void softmax_kernel(const float* __restrict__ pin,
                               _Float16* __restrict__ pout) {
  __shared__ float red[8];
  int row = blockIdx.x, tid = threadIdx.x;
  const float* pr = pin + (size_t)row * SS;
  float4 v = *(const float4*)(pr + tid * 4);
  float m = fmaxf(fmaxf(v.x, v.y), fmaxf(v.z, v.w));
#pragma unroll
  for (int o = 16; o; o >>= 1) m = fmaxf(m, __shfl_xor(m, o, 32));
  if ((tid & 31) == 0) red[tid >> 5] = m;
  __syncthreads();
  float rm = red[0];
#pragma unroll
  for (int i = 1; i < 8; ++i) rm = fmaxf(rm, red[i]);
  __syncthreads();
  float4 e;
  e.x = __expf(v.x - rm); e.y = __expf(v.y - rm);
  e.z = __expf(v.z - rm); e.w = __expf(v.w - rm);
  float s = e.x + e.y + e.z + e.w;
#pragma unroll
  for (int o = 16; o; o >>= 1) s += __shfl_xor(s, o, 32);
  if ((tid & 31) == 0) red[tid >> 5] = s;
  __syncthreads();
  float tot = 0.f;
#pragma unroll
  for (int i = 0; i < 8; ++i) tot += red[i];
  float inv = 1.f / tot;
  v4h o;
  o[0] = (_Float16)(e.x * inv); o[1] = (_Float16)(e.y * inv);
  o[2] = (_Float16)(e.z * inv); o[3] = (_Float16)(e.w * inv);
  *(v4h*)(pout + (size_t)row * SS + tid * 4) = o;
}

// MoE combine: rows routed to group g get  out = y_group + r_pre.
__global__ void moe_combine_kernel(const int* __restrict__ gid,
                                   const float* __restrict__ yg,
                                   const float* __restrict__ rp,
                                   float* __restrict__ out, int g) {
  int row = blockIdx.x;
  if (gid[row] != g) return;
  int tid = threadIdx.x;                       // 128 threads x float4 = 512
  size_t o = (size_t)row * DD + tid * 4;
  float4 a = *(const float4*)(yg + o);
  float4 b = *(const float4*)(rp + o);
  a.x += b.x; a.y += b.y; a.z += b.z; a.w += b.w;
  *(float4*)(out + o) = a;
}

// ---------------------------------------------------------------------------
// Host orchestration
// ---------------------------------------------------------------------------
struct AttnP {
  const float *bk,*bo,*bq,*bv,*conv_b,*conv_w,*ln1_b,*ln1_s,*ln2_b,*ln2_s,
              *wk,*wo,*wq,*wv;
};
struct PlainP {
  AttnP at;
  const float *b1,*b2,*ln3_b,*ln3_s,*w1,*w2;
};

extern "C" void kernel_launch(void* const* d_in, const int* in_sizes, int n_in,
                              void* d_out, int out_size, void* d_ws, size_t ws_size,
                              hipStream_t stream) {
  (void)in_sizes; (void)n_in; (void)out_size; (void)ws_size;
  auto F = [&](int i) { return (const float*)d_in[i]; };

  // Input order: top-level dict insertion order (x, group_ids, params);
  // params flattened jax-tree style (dict keys sorted, lists in order).
  const float* x_in  = F(0);
  const int*   gids  = (const int*)d_in[1];
  const float* ex_b1 = F(2);
  const float* ex_b2 = F(3);
  const float* ex_w1 = F(4);
  const float* ex_w2 = F(5);
  auto read_attn = [&](int i) {
    AttnP a = {F(i+0),F(i+1),F(i+2),F(i+3),F(i+4),F(i+5),F(i+6),F(i+7),
               F(i+8),F(i+9),F(i+10),F(i+11),F(i+12),F(i+13)};
    return a;
  };
  AttnP moe0 = read_attn(6);
  AttnP moe1 = read_attn(20);
  auto read_plain = [&](int i) {
    PlainP p;
    p.b1 = F(i+0);  p.b2 = F(i+1);
    p.at = {F(i+2),F(i+3),F(i+4),F(i+5),F(i+6),F(i+7),F(i+8),F(i+9),
            F(i+10),F(i+11),F(i+16),F(i+17),F(i+18),F(i+19)};
    p.ln3_b = F(i+12); p.ln3_s = F(i+13);
    p.w1 = F(i+14); p.w2 = F(i+15);
    return p;
  };
  PlainP plain0 = read_plain(34);
  PlainP plain1 = read_plain(54);

  // ---- workspace carve ----
  uint8_t* wsp = (uint8_t*)d_ws;
  auto carve = [&](size_t bytes) -> void* {
    void* p = wsp; wsp += (bytes + 255) & ~(size_t)255; return p;
  };
  const size_t ACTF = (size_t)MR * DD * sizeof(float);           // 8 MB
  const size_t ACTH = (size_t)MR * DD * sizeof(_Float16);        // 4 MB
  float* bufP   = (float*)carve(ACTF);                           // conv out / r_pre
  float* bufA   = (float*)carve(ACTF);                           // attn-block out
  float* bufR1  = (float*)carve(ACTF);                           // block outputs
  float* bufR2  = (float*)carve(ACTF);
  float* bufYG  = (float*)carve(ACTF);                           // MoE group accum
  float* probs  = (float*)carve((size_t)BB * HH * SS * SS * sizeof(float)); // 128 MB
  _Float16* bufLNh  = (_Float16*)carve(ACTH);
  _Float16* bufQh   = (_Float16*)carve(ACTH);
  _Float16* bufKh   = (_Float16*)carve(ACTH);
  _Float16* bufVh   = (_Float16*)carve(ACTH);
  _Float16* bufOh   = (_Float16*)carve(ACTH);
  _Float16* bufXh   = (_Float16*)carve(ACTH);
  _Float16* bufHidh = (_Float16*)carve((size_t)MR * FFD * sizeof(_Float16)); // 16 MB
  _Float16* probsh  = (_Float16*)carve((size_t)BB*HH*SS*SS*sizeof(_Float16)); // 64 MB
  _Float16* packW   = (_Float16*)carve((size_t)(TAPS*DD)*DD*sizeof(_Float16)); // 16 MB
  _Float16* packK   = (_Float16*)carve((size_t)BB*HH*(HDIM/32)*SS*32*sizeof(_Float16));
  _Float16* packV   = (_Float16*)carve((size_t)BB*HH*(SS/32)*HDIM*32*sizeof(_Float16));

  // ---- launch helpers ----
  auto run_ln = [&](const float* x, const float* s, const float* b, _Float16* y) {
    layernorm_kernel<<<dim3(MR), dim3(256), 0, stream>>>(x, s, b, y);
  };
  auto run_pack = [&](const float* W, int Kd, int N) {
    int64_t total = (int64_t)(Kd >> 5) * N;
    pack_b_kernel<<<dim3((unsigned)((total + 255) / 256)), dim3(256), 0, stream>>>(
        W, packW, Kd, N);
  };
  auto run_cvt = [&](const float* x, _Float16* y) {   // MR*DD elements
    cvt_f32_f16_kernel<<<dim3(MR * DD / 1024), dim3(256), 0, stream>>>(x, y);
  };
  auto run_gemm = [&](const _Float16* A, const _Float16* Bp, const float* bias,
                      const float* res, void* C, int M, int N, int Kd,
                      int lda, int ldc, float alpha, int act, int accum,
                      int outHalf, int Z, int Hdim,
                      int64_t a_sb, int64_t a_sh, int64_t b_sz,
                      int64_t c_sb, int64_t c_sh) {
    dim3 grid(N / 64, M / 128, Z), blk(32, 4);
    gemm_wmma_kernel<<<grid, blk, 0, stream>>>(A, Bp, bias, res, C, M, N, Kd,
                                               lda, ldc, a_sb, a_sh, b_sz,
                                               c_sb, c_sh, Hdim, alpha, act,
                                               accum, outHalf);
  };

  // conv + attention sub-blocks; returns x (post-attn) and r_pre (post-conv)
  auto conv_attn = [&](const float* xin, const AttnP& p, float** xout, float** rpre) {
    // conv sub-block: x = gelu(conv(ln1(x))) + x
    run_ln(xin, p.ln1_s, p.ln1_b, bufLNh);
    run_pack(p.conv_w, TAPS * DD, DD);
    conv_gemm_wmma_kernel<<<dim3(DD / 64, MR / 128), dim3(32, 4), 0, stream>>>(
        bufLNh, packW, p.conv_b, xin, bufP);
    // attention sub-block
    run_ln(bufP, p.ln2_s, p.ln2_b, bufLNh);
    run_pack(p.wq, DD, DD);
    run_gemm(bufLNh, packW, p.bq, nullptr, bufQh, MR, DD, DD, DD, DD,
             1.f, 0, 0, 1, 1, 1, 0, 0, 0, 0, 0);
    run_pack(p.wk, DD, DD);
    run_gemm(bufLNh, packW, p.bk, nullptr, bufKh, MR, DD, DD, DD, DD,
             1.f, 0, 0, 1, 1, 1, 0, 0, 0, 0, 0);
    run_pack(p.wv, DD, DD);
    run_gemm(bufLNh, packW, p.bv, nullptr, bufVh, MR, DD, DD, DD, DD,
             1.f, 0, 0, 1, 1, 1, 0, 0, 0, 0, 0);
    // scores = (Q.K^T)/8 per (b,h) -> f32, then softmax -> f16 probs
    pack_kT_kernel<<<dim3(8, BB * HH), dim3(256), 0, stream>>>(bufKh, packK);
    run_gemm(bufQh, packK, nullptr, nullptr, probs, SS, SS, HDIM, DD, SS,
             0.125f, 0, 0, 0, BB * HH, HH,
             (int64_t)SS * DD, HDIM, (int64_t)(HDIM / 32) * SS * 32,
             (int64_t)HH * SS * SS, (int64_t)SS * SS);
    softmax_kernel<<<dim3(BB * HH * SS), dim3(256), 0, stream>>>(probs, probsh);
    // O = P.V per (b,h) -> f16
    pack_v_kernel<<<dim3(8, BB * HH), dim3(256), 0, stream>>>(bufVh, packV);
    run_gemm(probsh, packV, nullptr, nullptr, bufOh, SS, HDIM, SS, SS, DD,
             1.f, 0, 0, 1, BB * HH, HH,
             (int64_t)HH * SS * SS, (int64_t)SS * SS,
             (int64_t)(SS / 32) * HDIM * 32, (int64_t)SS * DD, HDIM);
    // out-proj + residual (r_pre) -> f32
    run_pack(p.wo, DD, DD);
    run_gemm(bufOh, packW, p.bo, bufP, bufA, MR, DD, DD, DD, DD,
             1.f, 0, 0, 0, 1, 1, 0, 0, 0, 0, 0);
    *xout = bufA; *rpre = bufP;
  };

  auto plain_block = [&](const float* xin, const PlainP& p, float* outbuf) {
    float *x, *rp;
    conv_attn(xin, p.at, &x, &rp);
    run_ln(x, p.ln3_s, p.ln3_b, bufLNh);
    run_pack(p.w1, DD, FFD);
    run_gemm(bufLNh, packW, p.b1, nullptr, bufHidh, MR, FFD, DD, DD, FFD,
             1.f, 1 /*gelu*/, 0, 1, 1, 1, 0, 0, 0, 0, 0);
    run_pack(p.w2, FFD, DD);
    run_gemm(bufHidh, packW, p.b2, x /*residual*/, outbuf, MR, DD, FFD, FFD, DD,
             1.f, 0, 0, 0, 1, 1, 0, 0, 0, 0, 0);
  };

  auto moe_block = [&](const float* xin, const AttnP& p, float* outbuf) {
    float *x, *rp;
    conv_attn(xin, p, &x, &rp);
    run_cvt(x, bufXh);                     // xf in f16 for expert GEMMs
    for (int g = 0; g < GG; ++g) {
      for (int e = 0; e < EE; ++e) {
        int ge = g * EE + e;
        run_pack(ex_w1 + (size_t)ge * DD * HIDD, DD, HIDD);
        run_gemm(bufXh, packW, ex_b1 + (size_t)ge * HIDD, nullptr, bufHidh,
                 MR, HIDD, DD, DD, HIDD, 1.f, 1 /*gelu*/, 0, 1,
                 1, 1, 0, 0, 0, 0, 0);
        run_pack(ex_w2 + (size_t)ge * HIDD * DD, HIDD, DD);
        // mean over experts: 0.5*(h.w2 + b2), accumulate on e==1
        run_gemm(bufHidh, packW, ex_b2 + (size_t)ge * DD, nullptr, bufYG,
                 MR, DD, HIDD, HIDD, DD, 0.5f, 0, (e == 1) ? 1 : 0, 0,
                 1, 1, 0, 0, 0, 0, 0);
      }
      // rows routed to g: out = y_group + r_pre (pre-attention residual)
      moe_combine_kernel<<<dim3(MR), dim3(128), 0, stream>>>(gids, bufYG, rp,
                                                             outbuf, g);
    }
  };

  // ---- full model: plain0 -> moe0 -> moe1 -> plain1 ----
  plain_block(x_in, plain0, bufR1);
  moe_block(bufR1, moe0, bufR2);
  moe_block(bufR2, moe1, bufR1);
  plain_block(bufR1, plain1, (float*)d_out);
}
